// MultiHeadAttentionLayer_2989297238705
// MI455X (gfx1250) — matrix operand order repaired and checked by reading, not compiled, not run on hardware
//
#include <hip/hip_runtime.h>
#include <hip/hip_bf16.h>
#include <stdint.h>

// ---------------------------------------------------------------------------
// MultiHeadAttention for MI455X (gfx1250, wave32, WMMA bf16 -> f32 accum)
//   B=4, S=2048, D=1024, H=16, HD=64
// ---------------------------------------------------------------------------

typedef __attribute__((ext_vector_type(16))) __bf16 v16bf;
typedef __attribute__((ext_vector_type(8)))  __bf16 v8bf;
typedef __attribute__((ext_vector_type(8)))  float  v8f;
typedef __attribute__((ext_vector_type(4)))  float  v4f;
typedef __attribute__((ext_vector_type(4)))  int    v4i;

union ABFrag { v16bf v; v8bf h[2]; };

#define WMMA_BF16(A_, B_, C_) \
  __builtin_amdgcn_wmma_f32_16x16x32_bf16(false, (A_), false, (B_), (short)0, (C_), false, false)

// ---------------- async global->LDS (CDNA5 GLOBAL_LOAD_ASYNC_TO_LDS_B128) ---
#if __has_builtin(__builtin_amdgcn_global_load_async_to_lds_b128)
#define HAVE_ASYNC_LDS 1
static __device__ inline void async_cp16(const void* g, void* l) {
  __builtin_amdgcn_global_load_async_to_lds_b128(
      (__attribute__((address_space(1))) v4i*)(g),
      (__attribute__((address_space(3))) v4i*)(l), 0, 0);
}
#else
#define HAVE_ASYNC_LDS 0
static __device__ inline void async_cp16(const void* g, void* l) {
  *(v8bf*)l = *(const v8bf*)g;   // synchronous fallback
}
#endif
#define WAIT_ASYNCCNT(n) asm volatile("s_wait_asynccnt " #n ::: "memory")

// ---------------- width-16 XOR reductions via DPP16 row_xmask ---------------
#if __has_builtin(__builtin_amdgcn_update_dpp)
template <int CTRL>
static __device__ inline float dpp_xor(float x) {
  int i = __float_as_int(x);
  int s = __builtin_amdgcn_update_dpp(i, i, CTRL, 0xf, 0xf, true);
  return __int_as_float(s);
}
static __device__ inline float rowmax16(float x) {
  x = fmaxf(x, dpp_xor<0x161>(x));   // row_xmask:1
  x = fmaxf(x, dpp_xor<0x162>(x));   // row_xmask:2
  x = fmaxf(x, dpp_xor<0x164>(x));   // row_xmask:4
  x = fmaxf(x, dpp_xor<0x168>(x));   // row_xmask:8
  return x;
}
static __device__ inline float rowsum16(float x) {
  x += dpp_xor<0x161>(x);
  x += dpp_xor<0x162>(x);
  x += dpp_xor<0x164>(x);
  x += dpp_xor<0x168>(x);
  return x;
}
#else
static __device__ inline float rowmax16(float x) {
  x = fmaxf(x, __shfl_xor(x, 1, 16)); x = fmaxf(x, __shfl_xor(x, 2, 16));
  x = fmaxf(x, __shfl_xor(x, 4, 16)); x = fmaxf(x, __shfl_xor(x, 8, 16));
  return x;
}
static __device__ inline float rowsum16(float x) {
  x += __shfl_xor(x, 1, 16); x += __shfl_xor(x, 2, 16);
  x += __shfl_xor(x, 4, 16); x += __shfl_xor(x, 8, 16);
  return x;
}
#endif

// convert 8 contiguous fp32 -> 8 bf16 (v_cvt_pk_bf16_f32 pairs)
static __device__ inline v8bf cvt8(const float* __restrict__ p) {
  v4f x0 = *(const v4f*)p;
  v4f x1 = *(const v4f*)(p + 4);
  v8bf r;
  r[0] = (__bf16)x0[0]; r[1] = (__bf16)x0[1];
  r[2] = (__bf16)x0[2]; r[3] = (__bf16)x0[3];
  r[4] = (__bf16)x1[0]; r[5] = (__bf16)x1[1];
  r[6] = (__bf16)x1[2]; r[7] = (__bf16)x1[3];
  return r;
}

// ---------------------------------------------------------------------------
// GEMM: C[M,N] = A[M,K] * W[N,K]^T   (torch nn.Linear convention)
// MODE 0: bf16 row-major out            (Q/K projections)
// MODE 1: bf16 per-head-transposed out  (V projection -> Vt[b,h,hd,s])
// MODE 2: f32 + bias out                (output projection)
// ---------------------------------------------------------------------------
template <int MODE, typename AT>
__global__ __launch_bounds__(256) void mha_gemm_xwT(
    const AT* __restrict__ A, const float* __restrict__ W,
    void* __restrict__ outp, const float* __restrict__ bias,
    int M, int N, int K)
{
  const int lane = threadIdx.x & 31;
  const int wave = threadIdx.x >> 5;
  const int hi   = lane >> 4;
  const int ln   = lane & 15;
  const int wm   = wave & 3;
  const int wn   = wave >> 2;
  const int rowBase = blockIdx.y * 128 + wm * 32;
  const int colBase = blockIdx.x * 64  + wn * 32;

  v8f acc[2][2] = {};

  for (int k = 0; k < K; k += 32) {
    ABFrag a[2], b[2];
#pragma unroll
    for (int ms = 0; ms < 2; ++ms) {
      const AT* ap = A + (size_t)(rowBase + ms * 16 + ln) * K + k;
      if constexpr (sizeof(AT) == 2) {
        a[ms].h[0] = *(const v8bf*)((const __bf16*)ap + hi * 8);
        a[ms].h[1] = *(const v8bf*)((const __bf16*)ap + 16 + hi * 8);
      } else {
        a[ms].h[0] = cvt8((const float*)ap + hi * 8);
        a[ms].h[1] = cvt8((const float*)ap + 16 + hi * 8);
      }
    }
#pragma unroll
    for (int ns = 0; ns < 2; ++ns) {
      const float* wp = W + (size_t)(colBase + ns * 16 + ln) * K + k + hi * 8;
      b[ns].h[0] = cvt8(wp);
      b[ns].h[1] = cvt8(wp + 16);
    }
#pragma unroll
    for (int ms = 0; ms < 2; ++ms)
#pragma unroll
      for (int ns = 0; ns < 2; ++ns)
        acc[ms][ns] = WMMA_BF16(a[ms].v, b[ns].v, acc[ms][ns]);
  }

#pragma unroll
  for (int ms = 0; ms < 2; ++ms) {
#pragma unroll
    for (int ns = 0; ns < 2; ++ns) {
      const int rowT = rowBase + ms * 16 + hi * 8;
      const int col  = colBase + ns * 16 + ln;
#pragma unroll
      for (int g = 0; g < 8; ++g) {
        const int row = rowT + g;
        const float x = acc[ms][ns][g];
        if constexpr (MODE == 0) {
          ((__bf16*)outp)[(size_t)row * N + col] = (__bf16)x;
        } else if constexpr (MODE == 1) {
          const size_t dst = ((size_t)((row >> 11) * 16 + (col >> 6)) * 64 + (col & 63)) * 2048
                           + (size_t)(row & 2047);
          ((__bf16*)outp)[dst] = (__bf16)x;
        } else {
          ((float*)outp)[(size_t)row * N + col] = x + bias[col];
        }
      }
    }
  }
}

// ---------------------------------------------------------------------------
// Flash attention. Block = 8 waves = 128 query rows of one (b,h).
// K/V tiles double-buffered in LDS via async global->LDS copies; one wave owns
// 16 q-rows; online softmax with DPP row reductions; P transposed through LDS.
// All fragments of a phase are loaded into distinct registers BEFORE the WMMA
// chain so ds_loads clause together and overlap the matrix pipe.
// ---------------------------------------------------------------------------
__global__ __launch_bounds__(256) void mha_flash_attn(
    const __bf16* __restrict__ Qp, const __bf16* __restrict__ Kp,
    const __bf16* __restrict__ Vt, __bf16* __restrict__ Ctx)
{
  constexpr int S = 2048, D = 1024, HD = 64;
  constexpr int NIT = S / 32;
  constexpr float scale = 0.03125f; // 1/sqrt(1024)

  const int tid  = threadIdx.x;
  const int lane = tid & 31;
  const int wave = tid >> 5;
  const int hi   = lane >> 4;
  const int ln   = lane & 15;
  const int bh   = blockIdx.y;
  const int b    = bh >> 4;
  const int h    = bh & 15;
  const int q0   = blockIdx.x * 128 + wave * 16;

  // padded pitches: Kbuf 72 (144B = 9*16B), Vbuf 40 (80B = 5*16B) -> balanced banks
  __shared__ __align__(16) __bf16 Kbuf[2][32][72];
  __shared__ __align__(16) __bf16 Vbuf[2][64][40];
  __shared__ __align__(16) __bf16 Pbuf[8][16 * 32];
  __bf16* pb = Pbuf[wave];

  const __bf16* Kbase = Kp + (size_t)(b * S) * D + h * HD;
  const __bf16* Vbase = Vt + (size_t)bh * HD * S;

  // cooperative staging: K tile 32x64 (8 chunks/row), V tile 64x32 (4 chunks/row)
  const int kr = tid >> 3, kc = (tid & 7) * 8;
  const int vr = tid >> 2, vc = (tid & 3) * 8;
  auto issue_tile = [&](int k0, int nb) {
    async_cp16(Kbase + (size_t)(k0 + kr) * D + kc, &Kbuf[nb][kr][kc]);
    async_cp16(Vbase + (size_t)vr * S + k0 + vc,   &Vbuf[nb][vr][vc]);
  };

  // Q A-fragments, resident for the whole loop
  ABFrag qa[2];
  const __bf16* qrow = Qp + (size_t)(b * S + q0 + ln) * D + h * HD;
#pragma unroll
  for (int s = 0; s < 2; ++s) {
    qa[s].h[0] = *(const v8bf*)(qrow + s * 32 + hi * 16);
    qa[s].h[1] = *(const v8bf*)(qrow + s * 32 + hi * 16 + 8);
  }

  float mrow[8], lrow[8];
#pragma unroll
  for (int g = 0; g < 8; ++g) { mrow[g] = -1e30f; lrow[g] = 0.0f; }
  v8f O[4] = {};

  issue_tile(0, 0);

  for (int i = 0; i < NIT; ++i) {
    const int cb = i & 1;
    if (i + 1 < NIT) {               // prefetch next tile, then wait current
      issue_tile((i + 1) * 32, cb ^ 1);
      WAIT_ASYNCCNT(2);
    } else {
      WAIT_ASYNCCNT(0);
    }
    __syncthreads();                 // tile cb visible to all waves

    // ---- load ALL four K B-fragments first (one ds clause), then 4 WMMAs ---
    ABFrag kb[2][2];                 // [hd step][k col-tile]
#pragma unroll
    for (int s = 0; s < 2; ++s) {
      kb[s][0].h[0] = *(const v8bf*)&Kbuf[cb][ln][s * 32 + hi * 16];
      kb[s][0].h[1] = *(const v8bf*)&Kbuf[cb][ln][s * 32 + hi * 16 + 8];
      kb[s][1].h[0] = *(const v8bf*)&Kbuf[cb][16 + ln][s * 32 + hi * 16];
      kb[s][1].h[1] = *(const v8bf*)&Kbuf[cb][16 + ln][s * 32 + hi * 16 + 8];
    }
    v8f s0 = {}, s1 = {};
#pragma unroll
    for (int s = 0; s < 2; ++s) {
      s0 = WMMA_BF16(qa[s].v, kb[s][0].v, s0);
      s1 = WMMA_BF16(qa[s].v, kb[s][1].v, s1);
    }

    // ---- online softmax (rows striped as g + hi*8 across 16 lanes) ----
#pragma unroll
    for (int g = 0; g < 8; ++g) {
      float t0 = s0[g] * scale;
      float t1 = s1[g] * scale;
      const float mx   = rowmax16(fmaxf(t0, t1));
      const float mnew = fmaxf(mrow[g], mx);
      const float f    = __expf(mrow[g] - mnew);
      const float p0   = __expf(t0 - mnew);
      const float p1   = __expf(t1 - mnew);
      lrow[g] = lrow[g] * f + rowsum16(p0 + p1);
      mrow[g] = mnew;
      O[0][g] *= f; O[1][g] *= f; O[2][g] *= f; O[3][g] *= f;
      const int r = g + hi * 8;
      pb[r * 32 + ln]      = (__bf16)p0;
      pb[r * 32 + 16 + ln] = (__bf16)p1;
    }
    asm volatile("s_wait_dscnt 0" ::: "memory");

    // ---- load P A-fragment + ALL four V B-fragments, then 4 WMMAs ----
    ABFrag pa, vb[4];
    pa.h[0] = *(const v8bf*)(pb + ln * 32 + hi * 16);
    pa.h[1] = *(const v8bf*)(pb + ln * 32 + hi * 16 + 8);
#pragma unroll
    for (int t = 0; t < 4; ++t) {
      vb[t].h[0] = *(const v8bf*)&Vbuf[cb][t * 16 + ln][hi * 16];
      vb[t].h[1] = *(const v8bf*)&Vbuf[cb][t * 16 + ln][hi * 16 + 8];
    }
#pragma unroll
    for (int t = 0; t < 4; ++t)
      O[t] = WMMA_BF16(pa.v, vb[t].v, O[t]);

    __syncthreads();                 // done reading cb before it is refilled
  }

  // ---- epilogue: normalize, store context bf16 ----
#pragma unroll
  for (int g = 0; g < 8; ++g) {
    const float inv = 1.0f / lrow[g];
    const size_t row = (size_t)(b * S + q0 + g + hi * 8);
#pragma unroll
    for (int t = 0; t < 4; ++t)
      Ctx[row * D + h * HD + t * 16 + ln] = (__bf16)(O[t][g] * inv);
  }
}

// ---------------------------------------------------------------------------
extern "C" void kernel_launch(void* const* d_in, const int* in_sizes, int n_in,
                              void* d_out, int out_size, void* d_ws, size_t ws_size,
                              hipStream_t stream) {
  (void)in_sizes; (void)n_in; (void)out_size; (void)ws_size;
  const float* query = (const float*)d_in[0];
  const float* keys  = (const float*)d_in[1];
  const float* vals  = (const float*)d_in[2];
  const float* Wq    = (const float*)d_in[3];
  const float* Wk    = (const float*)d_in[4];
  const float* Wv    = (const float*)d_in[5];
  const float* Wo    = (const float*)d_in[6];
  const float* bo    = (const float*)d_in[7];

  constexpr int B = 4, S = 2048, D = 1024, H = 16;
  constexpr int M = B * S;
  constexpr size_t SLAB = (size_t)M * D * sizeof(__bf16);  // 16 MB

  char* ws = (char*)d_ws;
  __bf16* Qp  = (__bf16*)(ws + 0 * SLAB);
  __bf16* Kp  = (__bf16*)(ws + 1 * SLAB);
  __bf16* Vt  = (__bf16*)(ws + 2 * SLAB);        // [B*H, HD, S]
  __bf16* Ctx = (__bf16*)(ws + 3 * SLAB);

  dim3 blk(256);
  dim3 gproj(D / 64, M / 128);   // (16, 64)
  dim3 gattn(S / 128, B * H);    // (16, 64)

  mha_gemm_xwT<0, float><<<gproj, blk, 0, stream>>>(query, Wq, (void*)Qp, nullptr, M, D, D);
  mha_gemm_xwT<0, float><<<gproj, blk, 0, stream>>>(keys,  Wk, (void*)Kp, nullptr, M, D, D);
  mha_gemm_xwT<1, float><<<gproj, blk, 0, stream>>>(vals,  Wv, (void*)Vt, nullptr, M, D, D);
  mha_flash_attn<<<gattn, blk, 0, stream>>>(Qp, Kp, Vt, Ctx);
  mha_gemm_xwT<2, __bf16><<<gproj, blk, 0, stream>>>(Ctx, Wo, d_out, bo, M, D, D);
}
